// EfficientAttention_88330297410267
// MI455X (gfx1250) — matrix operand order repaired and checked by reading, not compiled, not run on hardware
//
#include <hip/hip_runtime.h>

// ---------------------------------------------------------------------------
// Multi-head attention forward for gfx1250 (MI455X), wave32 + WMMA bf16.
// B=2, S=2048, D=1024, H=16, hd=64.
// All matmuls run through v_wmma_f32_16x16x32_bf16 (f32 accumulate).
// V is produced head-transposed [B,H,hd,S] so every WMMA operand load in the
// attention core is a contiguous 128-bit load.
// ---------------------------------------------------------------------------

typedef unsigned short u16;
typedef unsigned int   u32;

#define B_   2
#define S_   2048
#define D_   1024
#define H_   16
#define HD_  64
#define M_   (B_ * S_)            // 4096 token rows

typedef __attribute__((ext_vector_type(16))) __bf16 v16bf;
typedef __attribute__((ext_vector_type(8)))  float  v8f;
typedef int v4i_ __attribute__((vector_size(16)));   // matches async-LDS builtin

// gfx1250 async global->LDS staging (guarded: falls back to plain copies)
#if defined(__HIP_DEVICE_COMPILE__) && defined(__gfx1250__)
#if __has_builtin(__builtin_amdgcn_global_load_async_to_lds_b128) && \
    __has_builtin(__builtin_amdgcn_s_wait_asynccnt)
#define HAVE_ASYNC_LDS 1
#endif
#endif
#ifndef HAVE_ASYNC_LDS
#define HAVE_ASYNC_LDS 0
#endif

struct BF16x16 {
    union { v16bf v; u32 u[8]; };
};

__device__ __forceinline__ u16 f32_to_bf16(float f) {
    u32 u = __float_as_uint(f);
    u32 r = (u + 0x7FFFu + ((u >> 16) & 1u)) >> 16;   // round to nearest even
    return (u16)r;
}

// A operand: 16x32 bf16 tile, rows = lanes 0..15 (mirrored in 16..31).
// VGPR j holds K pair: K = 16*half + 2*j (contiguous, matches load_b_nk)
__device__ __forceinline__ BF16x16 load_a(const u16* __restrict__ row0, int ld,
                                          int kb, int lane) {
    BF16x16 r;
    const int half = (lane >> 4) & 1;
    const u16* p = row0 + (size_t)(lane & 15) * ld + kb;
#pragma unroll
    for (int j = 0; j < 8; ++j) {
        int k0 = half * 16 + 2 * j;
        r.u[j] = *(const u32*)(p + k0);
    }
    return r;
}

// B operand from an [N,K] row-major matrix (torch [out,in] weights, K matrix,
// or head-transposed V): column n = lane&15, K pairs contiguous in memory.
// lanes 0-15: K 0..15, lanes 16-31: K 16..31 (VGPR v -> K = 16*half + 2v).
__device__ __forceinline__ BF16x16 load_b_nk(const u16* __restrict__ nrow0, int ld,
                                             int kb, int lane) {
    BF16x16 r;
    const int half = (lane >> 4) & 1;
    const u16* p = nrow0 + (size_t)(lane & 15) * ld + kb + half * 16;
#pragma unroll
    for (int v = 0; v < 8; ++v) r.u[v] = *(const u32*)(p + 2 * v);
    return r;
}

__device__ __forceinline__ v8f wmma_bf16(const BF16x16& a, const BF16x16& b, v8f c) {
    return __builtin_amdgcn_wmma_f32_16x16x32_bf16(
        /*neg_a=*/false, a.v, /*neg_b=*/false, b.v,
        /*c_mod=*/(short)0, c, /*reuse_a=*/false, /*reuse_b=*/false);
}

__device__ __forceinline__ float red_max16(float v) {
    v = fmaxf(v, __shfl_xor(v, 1));
    v = fmaxf(v, __shfl_xor(v, 2));
    v = fmaxf(v, __shfl_xor(v, 4));
    v = fmaxf(v, __shfl_xor(v, 8));
    return v;
}
__device__ __forceinline__ float red_sum16(float v) {
    v += __shfl_xor(v, 1);
    v += __shfl_xor(v, 2);
    v += __shfl_xor(v, 4);
    v += __shfl_xor(v, 8);
    return v;
}

// ---------------------------------------------------------------------------
// f32 -> bf16 conversion
// ---------------------------------------------------------------------------
__global__ void k_f32_to_bf16(const float* __restrict__ s, u16* __restrict__ d, int n) {
    int i = blockIdx.x * blockDim.x + threadIdx.x;
    if (i < n) d[i] = f32_to_bf16(s[i]);
}

// ---------------------------------------------------------------------------
// C[M,N] = A[M,K] @ W[N,K]^T + bias.  A bf16 [M,K] row-major, W bf16 [N,K]
// row-major (torch layout).  Output one of:
//   Cb : bf16 row-major [M,N]
//   Ct : bf16 head-transposed [B, N(=D), S]  (used for V)
//   Cf : f32  row-major [M,N]
// block = 128 threads (4 waves), each wave owns one 16-wide N tile.
// grid = (M/16, N/64).  K-loop is software-pipelined (double buffered regs).
// ---------------------------------------------------------------------------
__global__ void k_gemm_nt(const u16* __restrict__ A, const u16* __restrict__ W,
                          const float* __restrict__ bias,
                          u16* __restrict__ Cb, u16* __restrict__ Ct,
                          float* __restrict__ Cf,
                          int M, int N, int K) {
    const int tid  = threadIdx.x;
    const int lane = tid & 31;
    const int wave = tid >> 5;
    const int half = (lane >> 4) & 1;
    const int l16  = lane & 15;

    const int m0 = blockIdx.x * 16;
    const int n0 = blockIdx.y * 64 + wave * 16;

    const u16* arow0 = A + (size_t)m0 * K;
    const u16* wrow0 = W + (size_t)n0 * K;

    // software pipeline: operands for chunk k are loaded one iteration ahead
    BF16x16 a = load_a(arow0, K, 0, lane);
    BF16x16 b = load_b_nk(wrow0, K, 0, lane);
    v8f c = {};
    for (int kb = 32; kb < K; kb += 32) {
        BF16x16 a2 = load_a(arow0, K, kb, lane);
        BF16x16 b2 = load_b_nk(wrow0, K, kb, lane);
        c = wmma_bf16(a, b, c);
        a = a2; b = b2;
    }
    c = wmma_bf16(a, b, c);

    const int col = n0 + l16;
    const float bv = bias[col];

    if (Ct) {
        // head-transposed store: the 8 accumulator rows of this lane are the
        // contiguous s-range [s0, s0+8) of column `col` -> one b128 store.
        const int bb = m0 / S_;
        const int s0 = (m0 % S_) + 8 * half;
        u32 w[4];
#pragma unroll
        for (int r = 0; r < 4; ++r) {
            u32 lo = f32_to_bf16(c[2 * r] + bv);
            u32 hi = f32_to_bf16(c[2 * r + 1] + bv);
            w[r] = lo | (hi << 16);
        }
        u32* dst = (u32*)(Ct + ((size_t)bb * D_ + col) * S_ + s0);
        dst[0] = w[0]; dst[1] = w[1]; dst[2] = w[2]; dst[3] = w[3];
    } else {
#pragma unroll
        for (int r = 0; r < 8; ++r) {
            const int m = m0 + r + 8 * half;
            const float val = c[r] + bv;
            if (Cf) Cf[(size_t)m * N + col] = val;
            else    Cb[(size_t)m * N + col] = f32_to_bf16(val);
        }
    }
}

// ---------------------------------------------------------------------------
// Fused attention core: one workgroup per (b, h, 16-row q tile).
// 256 threads = 8 waves.  Dynamic LDS:
//   [0)       f32 scores [16][S]                 131072 B
//   [131072)  bf16 P     [16][S]                  65536 B
//   [196608)  f32 partial attended [2][16][HD]     8192 B
//   [204800)  bf16 Q tile [16][HD]                 2048 B
// Vt is head-transposed V: [B, D(=h*HD+d), S] bf16.
// ---------------------------------------------------------------------------
__global__ void k_attention(const u16* __restrict__ Qb, const u16* __restrict__ Kb,
                            const u16* __restrict__ Vt,
                            float* __restrict__ attn, u16* __restrict__ attended) {
    extern __shared__ char smem[];
    float* s_scores = (float*)smem;                               // [16][S_]
    u16*   s_p      = (u16*)(smem + 16 * S_ * 4);                 // [16][S_]
    float* s_att    = (float*)(smem + 16 * S_ * 4 + 16 * S_ * 2); // [2][16][HD_]
    u16*   s_q      = (u16*)(smem + 16 * S_ * 4 + 16 * S_ * 2
                                  + 2 * 16 * HD_ * 4);            // [16][HD_]

    const int qt = blockIdx.x;   // q tile (16 rows)
    const int h  = blockIdx.y;
    const int b  = blockIdx.z;

    const int tid  = threadIdx.x;
    const int lane = tid & 31;
    const int wave = tid >> 5;
    const int half = (lane >> 4) & 1;
    const int l16  = lane & 15;

    const size_t qrow0 = (size_t)b * S_ + (size_t)qt * 16;
    const int hoff = h * HD_;

    // ---- stage Q tile (16 x 64 bf16 = 2 KB) into LDS, async if available
    if (tid < 128) {
        const int row = tid >> 3, seg = tid & 7;               // 8 u16 per thread
        const u16* g = Qb + (qrow0 + row) * D_ + hoff + seg * 8;
        u16* l = s_q + row * HD_ + seg * 8;
#if HAVE_ASYNC_LDS
        __builtin_amdgcn_global_load_async_to_lds_b128(
            (__attribute__((address_space(1))) v4i_*)g,
            (__attribute__((address_space(3))) v4i_*)l, 0, 0);
#else
        const u32* gs = (const u32*)g;
        u32* ls = (u32*)l;
        ls[0] = gs[0]; ls[1] = gs[1]; ls[2] = gs[2]; ls[3] = gs[3];
#endif
    }
#if HAVE_ASYNC_LDS
    __builtin_amdgcn_s_wait_asynccnt(0);
#endif
    __syncthreads();

    // ---- phase 1: scores = (Q K^T) / sqrt(hd), key tiles split across 8 waves
    BF16x16 a0 = load_a(s_q, HD_, 0, lane);
    BF16x16 a1 = load_a(s_q, HD_, 32, lane);
    for (int nt = wave; nt < S_ / 16; nt += 8) {
        const u16* krow0 = Kb + ((size_t)b * S_ + (size_t)nt * 16) * D_ + hoff;
        BF16x16 b0 = load_b_nk(krow0, D_, 0, lane);
        BF16x16 b1 = load_b_nk(krow0, D_, 32, lane);
        v8f c = {};
        c = wmma_bf16(a0, b0, c);
        c = wmma_bf16(a1, b1, c);
#pragma unroll
        for (int r = 0; r < 8; ++r) {
            int m = r + 8 * half;
            s_scores[m * S_ + nt * 16 + l16] = c[r] * 0.125f;  // 1/sqrt(64)
        }
    }
    __syncthreads();

    // ---- phase 2: softmax, 16 threads per row; emit f32 weights + bf16 P
    {
        const int r  = tid >> 4;
        const int c0 = tid & 15;
        float* row = s_scores + r * S_;
        float mx = -3.402823e38f;
        for (int i = c0; i < S_; i += 16) mx = fmaxf(mx, row[i]);
        mx = red_max16(mx);
        float sum = 0.f;
        for (int i = c0; i < S_; i += 16) {
            float e = __expf(row[i] - mx);
            row[i] = e;
            sum += e;
        }
        sum = red_sum16(sum);
        const float inv = 1.0f / sum;
        const size_t obase =
            (((size_t)b * H_ + h) * S_ + (size_t)(qt * 16 + r)) * S_;
        u16* prow = s_p + r * S_;
        for (int i = c0; i < S_; i += 16) {
            float p = row[i] * inv;
            attn[obase + i] = p;
            prow[i] = f32_to_bf16(p);
        }
    }
    __syncthreads();

    // ---- phase 3: attended = P @ V.  8 waves = 4 N tiles x 2 K halves.
    // B operand reads head-transposed V -> contiguous b128 loads.
    {
        const int ntile = wave & 3;
        const int khalf = wave >> 2;
        const u16* vrow0 = Vt + ((size_t)b * D_ + hoff + ntile * 16) * S_;
        const int kbase = khalf * 1024;

        BF16x16 a  = load_a(s_p, S_, kbase, lane);
        BF16x16 bb = load_b_nk(vrow0, S_, kbase, lane);
        v8f c = {};
        for (int kc = 1; kc < 32; ++kc) {
            const int kb = kbase + kc * 32;
            BF16x16 a2  = load_a(s_p, S_, kb, lane);
            BF16x16 bb2 = load_b_nk(vrow0, S_, kb, lane);
            c = wmma_bf16(a, bb, c);
            a = a2; bb = bb2;
        }
        c = wmma_bf16(a, bb, c);
#pragma unroll
        for (int r = 0; r < 8; ++r) {
            int m = r + 8 * half;
            s_att[(khalf * 16 + m) * HD_ + ntile * 16 + l16] = c[r];
        }
    }
    __syncthreads();

    for (int idx = tid; idx < 16 * HD_; idx += 256) {
        const int m = idx / HD_, j = idx % HD_;
        float v = s_att[m * HD_ + j] + s_att[(16 + m) * HD_ + j];
        attended[(qrow0 + m) * D_ + hoff + j] = f32_to_bf16(v);
    }
}

// ---------------------------------------------------------------------------
// Host-side launch
// ---------------------------------------------------------------------------
extern "C" void kernel_launch(void* const* d_in, const int* in_sizes, int n_in,
                              void* d_out, int out_size, void* d_ws, size_t ws_size,
                              hipStream_t stream) {
    (void)in_sizes; (void)n_in; (void)out_size; (void)ws_size;

    const float* q_in = (const float*)d_in[0];
    const float* k_in = (const float*)d_in[1];
    const float* v_in = (const float*)d_in[2];
    const float* Wq   = (const float*)d_in[3];
    const float* bq   = (const float*)d_in[4];
    const float* Wk   = (const float*)d_in[5];
    const float* bk   = (const float*)d_in[6];
    const float* Wv   = (const float*)d_in[7];
    const float* bv   = (const float*)d_in[8];
    const float* Wo   = (const float*)d_in[9];
    const float* bo   = (const float*)d_in[10];

    float* out  = (float*)d_out;                       // [B,S,D] fp32
    float* attn = out + (size_t)M_ * D_;               // [B,H,S,S] fp32

    // workspace carve-up (all bf16 buffers as u16)
    char* ws = (char*)d_ws;
    size_t off = 0;
    auto carve = [&](size_t bytes) -> char* {
        char* p = ws + off;
        off = (off + bytes + 255) & ~(size_t)255;
        return p;
    };
    const size_t MD = (size_t)M_ * D_;
    const size_t DD = (size_t)D_ * D_;
    u16* Xq  = (u16*)carve(MD * 2);
    u16* Xk  = (u16*)carve(MD * 2);
    u16* Xv  = (u16*)carve(MD * 2);
    u16* Wqb = (u16*)carve(DD * 2);
    u16* Wkb = (u16*)carve(DD * 2);
    u16* Wvb = (u16*)carve(DD * 2);
    u16* Wob = (u16*)carve(DD * 2);
    u16* Qb  = (u16*)carve(MD * 2);
    u16* Kb  = (u16*)carve(MD * 2);
    u16* Vt  = (u16*)carve(MD * 2);   // head-transposed V: [B, D, S]
    u16* Att = (u16*)carve(MD * 2);

    // 1) convert activations + weights to bf16
    {
        const int n1 = (int)MD, g1 = (n1 + 255) / 256;
        k_f32_to_bf16<<<g1, 256, 0, stream>>>(q_in, Xq, n1);
        k_f32_to_bf16<<<g1, 256, 0, stream>>>(k_in, Xk, n1);
        k_f32_to_bf16<<<g1, 256, 0, stream>>>(v_in, Xv, n1);
        const int n2 = (int)DD, g2 = (n2 + 255) / 256;
        k_f32_to_bf16<<<g2, 256, 0, stream>>>(Wq, Wqb, n2);
        k_f32_to_bf16<<<g2, 256, 0, stream>>>(Wk, Wkb, n2);
        k_f32_to_bf16<<<g2, 256, 0, stream>>>(Wv, Wvb, n2);
        k_f32_to_bf16<<<g2, 256, 0, stream>>>(Wo, Wob, n2);
    }

    // 2) Q/K/V projections (Q,K row-major bf16; V head-transposed bf16)
    {
        dim3 grid(M_ / 16, D_ / 64);
        k_gemm_nt<<<grid, 128, 0, stream>>>(Xq, Wqb, bq, Qb, nullptr, nullptr,
                                            M_, D_, D_);
        k_gemm_nt<<<grid, 128, 0, stream>>>(Xk, Wkb, bk, Kb, nullptr, nullptr,
                                            M_, D_, D_);
        k_gemm_nt<<<grid, 128, 0, stream>>>(Xv, Wvb, bv, nullptr, Vt, nullptr,
                                            M_, D_, D_);
    }

    // 3) fused attention (scores -> softmax -> P@V)
    {
        dim3 grid(S_ / 16, H_, B_);
        const size_t lds = (size_t)16 * S_ * 4   // f32 scores
                         + (size_t)16 * S_ * 2   // bf16 P
                         + (size_t)2 * 16 * HD_ * 4  // partial attended
                         + (size_t)16 * HD_ * 2; // Q tile
        k_attention<<<grid, 256, lds, stream>>>(Qb, Kb, Vt, attn, Att);
    }

    // 4) output projection (f32 out -> d_out)
    {
        dim3 grid(M_ / 16, D_ / 64);
        k_gemm_nt<<<grid, 128, 0, stream>>>(Att, Wob, bo, nullptr, nullptr, out,
                                            M_, D_, D_);
    }
}
